// Attention_37297495999189
// MI455X (gfx1250) — matrix operand order repaired and checked by reading, not compiled, not run on hardware
//
#include <hip/hip_runtime.h>
#include <hip/hip_bf16.h>
#include <math.h>

// ---------------------------------------------------------------------------
// Problem constants: B=8, N=1024, C=768, H=12, R=64, HD=64
// ---------------------------------------------------------------------------
#define BATCH   8
#define SEQ     1024
#define CH      768
#define NHEAD   12
#define HD      64
#define THREEC  (3 * CH)         // 2304
#define ROWS    (BATCH * SEQ)    // 8192
#define NROUT   (2 * NHEAD * 3)  // 72

typedef __attribute__((ext_vector_type(16))) _Float16 v16h;
typedef __attribute__((ext_vector_type(8)))  _Float16 v8h;
typedef __attribute__((ext_vector_type(8)))  float    v8f;

__device__ __forceinline__ v8f wmma_f16(const v16h& a, const v16h& b, const v8f& c) {
  return __builtin_amdgcn_wmma_f32_16x16x32_f16(false, a, false, b, (short)0, c, false, false);
}

// CDNA5 async global->LDS (ASYNCcnt-tracked, 16B per lane).
// lds_off is the byte offset from the workgroup LDS base (kernel uses only
// dynamic LDS, so dynamic-smem element offsets are LDS offsets directly).
__device__ __forceinline__ void async_b128(unsigned lds_off, const _Float16* g) {
  asm volatile("global_load_async_to_lds_b128 %0, %1, off"
               :: "v"(lds_off), "v"(g) : "memory");
}
__device__ __forceinline__ void wait_async0() {
  asm volatile("s_wait_asynccnt 0x0" ::: "memory");
}

// ---------------------------------------------------------------------------
// fp32 -> fp16 convert with row padding (pad rows -> 0) so GEMM B-tiles need
// no bounds checks and async loads are unconditional.
// ---------------------------------------------------------------------------
__global__ __launch_bounds__(256)
void cvt_f16(const float* __restrict__ src, _Float16* __restrict__ dst,
             int rowsValid, int K, int total)
{
  const int i = blockIdx.x * 256 + threadIdx.x;
  if (i >= total) return;
  const int r = i / K;
  dst[i] = (r < rowsValid) ? (_Float16)src[i] : (_Float16)0.0f;
}

// ---------------------------------------------------------------------------
// f16 WMMA GEMM with double-buffered async-LDS staging:
//   Out[M,Nout] = A16[M,K] @ W16[NoutPad,K]^T  (+ epilogue)
// Block tile 64(M) x 256(N), K-step 32; 8 waves (2x4), wave tile 32x64
// (8 WMMA accumulators). LDS: 2 x (64*32 + 256*32) halves = 40KB dynamic.
// Modes:
//   0: OutH = acc (f16)
//   1: Out = acc*p0[n] + p1[n]
//   2: OutH = gelu(acc + p0[n]) (f16)
//   3: Out = sigmoid(acc + p0[n])
//   4: Out[r,n] += acc * aux2[r*72 + 2*(3h+s)]
//   5: Out[r,n] += (acc+p0[n]) * (s ? aux2[r*72+2*(3h+s)+1] : 0)
//   6: Out = (acc + p0[n])*p1[n] + aux2[n]
// ---------------------------------------------------------------------------
#define GTILE (64 * 32 + 256 * 32)   // halves per LDS buffer (A then B)

__global__ __launch_bounds__(256)
void gemm_wmma_f16(const _Float16* __restrict__ A, const _Float16* __restrict__ W,
                   float* __restrict__ Out, _Float16* __restrict__ OutH,
                   int M, int Nout, int K, int mode,
                   const float* __restrict__ p0, const float* __restrict__ p1,
                   const float* __restrict__ aux2)
{
  extern __shared__ __align__(16) _Float16 smem[];   // only dynamic LDS here

  const int tid  = threadIdx.x;
  const int lane = tid & 31;
  const int wid  = tid >> 5;
  const int wm   = wid & 1;        // 2 waves in M
  const int wn   = wid >> 1;       // 4 waves in N
  const int grp  = lane >> 4;
  const int l16  = lane & 15;
  const int m0   = blockIdx.y * 64;
  const int n0   = blockIdx.x * 256;

  // stage one K-slab into LDS buffer `buf` via async copies
  auto stageA = [&](int k0, int buf) {
    const int r = tid >> 2, c = (tid & 3) * 8;               // 64x32: 1 chunk/thread
    async_b128((unsigned)((buf * GTILE + r * 32 + c) * 2),
               A + (size_t)(m0 + r) * K + k0 + c);
  };
  auto stageB = [&](int k0, int buf) {
    #pragma unroll
    for (int i = 0; i < 4; ++i) {                            // 256x32: 4 chunks/thread
      const int ci = i * 256 + tid;
      const int r = ci >> 2, c = (ci & 3) * 8;
      async_b128((unsigned)((buf * GTILE + 2048 + r * 32 + c) * 2),
                 W + (size_t)(n0 + r) * K + k0 + c);
    }
  };

  v8f acc[2][4];
  #pragma unroll
  for (int mi = 0; mi < 2; ++mi)
    #pragma unroll
    for (int ni = 0; ni < 4; ++ni)
      #pragma unroll
      for (int j = 0; j < 8; ++j) acc[mi][ni][j] = 0.0f;

  stageA(0, 0);
  stageB(0, 0);
  int cur = 0;

  for (int k0 = 0; k0 < K; k0 += 32) {
    wait_async0();          // my async writes (to buffer `cur`) are done
    __syncthreads();        // everyone's are done; prior reads of cur^1 done
    if (k0 + 32 < K) {      // prefetch next slab into the other buffer
      stageA(k0 + 32, cur ^ 1);
      stageB(k0 + 32, cur ^ 1);
    }
    const _Float16* As = smem + cur * GTILE;
    const _Float16* Bs = As + 2048;

    // A-fragments (16-bit 16x32): lane l16 = row, 16 contiguous K at grp*16
    v16h af[2];
    #pragma unroll
    for (int mi = 0; mi < 2; ++mi) {
      const int r = wm * 32 + mi * 16 + l16;
      const v8h lo = *(const v8h*)(As + r * 32 + grp * 16);
      const v8h hi = *(const v8h*)(As + r * 32 + grp * 16 + 8);
      #pragma unroll
      for (int i = 0; i < 8; ++i) { af[mi][i] = lo[i]; af[mi][i + 8] = hi[i]; }
    }
    // B-fragments (16-bit 32x16): lane l16 = col, 16 contiguous K at grp*16
    v16h bf[4];
    #pragma unroll
    for (int ni = 0; ni < 4; ++ni) {
      const int r = wn * 64 + ni * 16 + l16;
      bf[ni] = *(const v16h*)(Bs + r * 32 + grp * 16);
    }
    #pragma unroll
    for (int mi = 0; mi < 2; ++mi)
      #pragma unroll
      for (int ni = 0; ni < 4; ++ni)
        acc[mi][ni] = wmma_f16(af[mi], bf[ni], acc[mi][ni]);
    cur ^= 1;
  }

  // --- epilogue ---
  #pragma unroll
  for (int mi = 0; mi < 2; ++mi) {
    #pragma unroll
    for (int ni = 0; ni < 4; ++ni) {
      const int n = n0 + wn * 64 + ni * 16 + l16;
      if (n >= Nout) continue;
      const int s = n / CH;
      const int h = (n % CH) >> 6;
      #pragma unroll
      for (int j = 0; j < 8; ++j) {
        const int row = m0 + wm * 32 + mi * 16 + grp * 8 + j;
        if (row >= M) continue;
        float v = acc[mi][ni][j];
        const size_t oidx = (size_t)row * Nout + n;
        switch (mode) {
          case 0:
            OutH[oidx] = (_Float16)v; break;
          case 1:
            Out[oidx] = v * p0[n] + p1[n]; break;
          case 2: {
            v += p0[n];
            OutH[oidx] = (_Float16)(0.5f * v * (1.0f + erff(v * 0.70710678118654752f)));
            break;
          }
          case 3: {
            v += p0[n];
            Out[oidx] = 1.0f / (1.0f + __expf(-v)); break;
          }
          case 4: {
            const float g = aux2[(size_t)row * NROUT + 2 * (3 * h + s)];
            Out[oidx] += v * g; break;
          }
          case 5: {
            v += p0[n];
            const float g = (s == 0) ? 0.0f
                          : aux2[(size_t)row * NROUT + 2 * (3 * h + s) + 1];
            Out[oidx] += v * g; break;
          }
          case 6:
            Out[oidx] = (v + p0[n]) * p1[n] + aux2[n]; break;
        }
      }
    }
  }
}

// ---------------------------------------------------------------------------
// Repack gated QKV (fp32 [8192,2304]) -> f16 attention operands:
//   qh [BH,N,64] (pre-scaled by HD^-0.5=0.125), kh [BH,N,64], vT [BH,64,N]
// ---------------------------------------------------------------------------
__global__ __launch_bounds__(256)
void pack_qkv(const float* __restrict__ buf,
              _Float16* __restrict__ qh, _Float16* __restrict__ kh,
              _Float16* __restrict__ vT)
{
  const int idx = blockIdx.x * 256 + threadIdx.x;     // over 8192*768
  if (idx >= ROWS * CH) return;
  const int c   = idx % CH;
  const int row = idx / CH;
  const int n   = row % SEQ;
  const int b   = row / SEQ;
  const int d   = c & 63;
  const int h   = c >> 6;
  const int bh  = b * NHEAD + h;
  const float qv = buf[(size_t)row * THREEC + c];
  const float kv = buf[(size_t)row * THREEC + CH + c];
  const float vv = buf[(size_t)row * THREEC + 2 * CH + c];
  qh[((size_t)bh * SEQ + n) * HD + d] = (_Float16)(qv * 0.125f);
  kh[((size_t)bh * SEQ + n) * HD + d] = (_Float16)kv;
  vT[((size_t)bh * HD + d) * SEQ + n] = (_Float16)vv;
}

// ---------------------------------------------------------------------------
// Flash-style attention: 4 waves x 16 query rows; 32-key-column steps.
// S (16x32) = 4 WMMA; online softmax via padded LDS; P->A-fragment via LDS;
// O (16x64) update = 4 WMMA with per-row alpha rescale. Output f16.
// ---------------------------------------------------------------------------
__global__ __launch_bounds__(128)
void attn_wmma(const _Float16* __restrict__ qh, const _Float16* __restrict__ kh,
               const _Float16* __restrict__ vT, _Float16* __restrict__ o)
{
  __shared__ __align__(16) float     Sbuf[4][16 * 33];
  __shared__ __align__(16) _Float16  Pbuf[4][16 * 32];
  __shared__ __align__(16) float     Rbuf[4][16];

  const int tid  = threadIdx.x;
  const int lane = tid & 31;
  const int wid  = tid >> 5;
  const int grp  = lane >> 4;
  const int l16  = lane & 15;
  const int bh   = blockIdx.x >> 4;
  const int rb   = blockIdx.x & 15;
  const int row0 = rb * 64 + wid * 16;

  const _Float16* qrow = qh + ((size_t)bh * SEQ + row0 + l16) * HD;
  v16h qf[2];
  #pragma unroll
  for (int kk = 0; kk < 2; ++kk) {
    const v8h lo = *(const v8h*)(qrow + kk * 32 + grp * 16);
    const v8h hi = *(const v8h*)(qrow + kk * 32 + grp * 16 + 8);
    #pragma unroll
    for (int i = 0; i < 8; ++i) { qf[kk][i] = lo[i]; qf[kk][i + 8] = hi[i]; }
  }

  float run_m = -INFINITY, run_s = 0.0f;
  v8f oacc[4];
  #pragma unroll
  for (int t = 0; t < 4; ++t)
    #pragma unroll
    for (int j = 0; j < 8; ++j) oacc[t][j] = 0.0f;

  for (int nb = 0; nb < SEQ; nb += 32) {
    v8f s[2];
    #pragma unroll
    for (int ni = 0; ni < 2; ++ni) {
      #pragma unroll
      for (int j = 0; j < 8; ++j) s[ni][j] = 0.0f;
      const _Float16* krow = kh + ((size_t)bh * SEQ + nb + ni * 16 + l16) * HD;
      #pragma unroll
      for (int kk = 0; kk < 2; ++kk) {
        const v16h kf = *(const v16h*)(krow + kk * 32 + grp * 16);
        s[ni] = wmma_f16(qf[kk], kf, s[ni]);
      }
    }
    __syncthreads();
    #pragma unroll
    for (int ni = 0; ni < 2; ++ni)
      #pragma unroll
      for (int j = 0; j < 8; ++j)
        Sbuf[wid][(grp * 8 + j) * 33 + ni * 16 + l16] = s[ni][j];
    __syncthreads();

    if (lane < 16) {
      const float* Sr = &Sbuf[wid][lane * 33];
      float tmax = Sr[0];
      #pragma unroll 4
      for (int c = 1; c < 32; ++c) tmax = fmaxf(tmax, Sr[c]);
      const float nm    = fmaxf(run_m, tmax);
      const float alpha = __expf(run_m - nm);
      float sum = 0.0f;
      _Float16* Pr = &Pbuf[wid][lane * 32];
      #pragma unroll 4
      for (int c = 0; c < 32; ++c) {
        const float p = __expf(Sr[c] - nm);
        Pr[c] = (_Float16)p;
        sum += p;
      }
      run_s = run_s * alpha + sum;
      run_m = nm;
      Rbuf[wid][lane] = alpha;
    }
    __syncthreads();

    v16h pf;
    {
      const v8h lo = *(const v8h*)(&Pbuf[wid][l16 * 32 + grp * 16]);
      const v8h hi = *(const v8h*)(&Pbuf[wid][l16 * 32 + grp * 16 + 8]);
      #pragma unroll
      for (int i = 0; i < 8; ++i) { pf[i] = lo[i]; pf[i + 8] = hi[i]; }
    }
    float al[8];
    #pragma unroll
    for (int j = 0; j < 8; ++j) al[j] = Rbuf[wid][grp * 8 + j];
    #pragma unroll
    for (int t = 0; t < 4; ++t) {
      v8f c;
      #pragma unroll
      for (int j = 0; j < 8; ++j) c[j] = oacc[t][j] * al[j];
      const v16h vf = *(const v16h*)(vT + ((size_t)bh * HD + t * 16 + l16) * SEQ
                                     + nb + grp * 16);
      oacc[t] = wmma_f16(pf, vf, c);
    }
  }

  __syncthreads();
  if (lane < 16) Rbuf[wid][lane] = 1.0f / run_s;
  __syncthreads();

  const int b = bh / NHEAD, h = bh % NHEAD;
  #pragma unroll
  for (int t = 0; t < 4; ++t)
    #pragma unroll
    for (int j = 0; j < 8; ++j) {
      const int m    = grp * 8 + j;
      const int nrow = row0 + m;
      o[((size_t)(b * SEQ + nrow)) * CH + h * HD + t * 16 + l16] =
          (_Float16)(oacc[t][j] * Rbuf[wid][m]);
    }
}

// ---------------------------------------------------------------------------
// Orchestration (graph-capture safe: launches only)
// ---------------------------------------------------------------------------
extern "C" void kernel_launch(void* const* d_in, const int* in_sizes, int n_in,
                              void* d_out, int out_size, void* d_ws, size_t ws_size,
                              hipStream_t stream) {
  const float* x          = (const float*)d_in[0];
  const float* Wqkv       = (const float*)d_in[1];
  const float* ssf_scale1 = (const float*)d_in[2];
  const float* ssf_shift1 = (const float*)d_in[3];
  const float* Wa         = (const float*)d_in[4];
  const float* Wb         = (const float*)d_in[5];
  const float* Wrouter    = (const float*)d_in[6];
  const float* brouter    = (const float*)d_in[7];
  const float* Wad_down   = (const float*)d_in[8];
  const float* bad_down   = (const float*)d_in[9];
  const float* Wad_up     = (const float*)d_in[10];
  const float* bad_up     = (const float*)d_in[11];
  const float* Wproj      = (const float*)d_in[12];
  const float* bproj      = (const float*)d_in[13];
  const float* ssf_scale2 = (const float*)d_in[14];
  const float* ssf_shift2 = (const float*)d_in[15];
  float* out = (float*)d_out;

  char* ws = (char*)d_ws;
  size_t off = 0;
  auto carve = [&](size_t bytes) -> void* {
    void* p = ws + off;
    off = (off + bytes + 255) & ~(size_t)255;
    return p;
  };
  // fp32 intermediates
  float*    buf_qkv = (float*)   carve((size_t)ROWS * THREEC * 4);   // 75.5 MB
  float*    router  = (float*)   carve((size_t)ROWS * NROUT * 4);
  // f16 operands
  _Float16* x16     = (_Float16*)carve((size_t)ROWS * CH * 2);
  _Float16* Wqkv16  = (_Float16*)carve((size_t)THREEC * CH * 2);
  _Float16* Wa16    = (_Float16*)carve((size_t)256 * CH * 2);
  _Float16* Wadd16  = (_Float16*)carve((size_t)256 * CH * 2);
  _Float16* Wrt16   = (_Float16*)carve((size_t)256 * CH * 2);
  _Float16* Wb16    = (_Float16*)carve((size_t)THREEC * 64 * 2);
  _Float16* Wadu16  = (_Float16*)carve((size_t)THREEC * 64 * 2);
  _Float16* Wproj16 = (_Float16*)carve((size_t)CH * CH * 2);
  _Float16* t_a16   = (_Float16*)carve((size_t)ROWS * 64 * 2);
  _Float16* t_ad16  = (_Float16*)carve((size_t)ROWS * 64 * 2);
  _Float16* o16     = (_Float16*)carve((size_t)ROWS * CH * 2);
  _Float16* q16     = (_Float16*)carve((size_t)BATCH * NHEAD * SEQ * HD * 2);
  _Float16* k16     = (_Float16*)carve((size_t)BATCH * NHEAD * SEQ * HD * 2);
  _Float16* vT16    = (_Float16*)carve((size_t)BATCH * NHEAD * SEQ * HD * 2);
  (void)ws_size; (void)in_sizes; (void)n_in; (void)out_size;

  const dim3 blk(256);
  const int mtiles = ROWS / 64;          // 128
  const size_t gemm_lds = (size_t)2 * GTILE * sizeof(_Float16);  // 40 KB

  auto cvt = [&](const float* s, _Float16* d, int rowsValid, int K, int rowsPad) {
    const int total = rowsPad * K;
    cvt_f16<<<dim3((total + 255) / 256), blk, 0, stream>>>(s, d, rowsValid, K, total);
  };
  // one-time f16 conversions (padded rows zeroed)
  cvt(x,        x16,     ROWS,   CH, ROWS);
  cvt(Wqkv,     Wqkv16,  THREEC, CH, THREEC);
  cvt(Wa,       Wa16,    64,     CH, 256);
  cvt(Wad_down, Wadd16,  64,     CH, 256);
  cvt(Wrouter,  Wrt16,   NROUT,  CH, 256);
  cvt(Wb,       Wb16,    THREEC, 64, THREEC);
  cvt(Wad_up,   Wadu16,  THREEC, 64, THREEC);
  cvt(Wproj,    Wproj16, CH,     CH, CH);

  // 1) base QKV + SSF
  gemm_wmma_f16<<<dim3(THREEC / 256, mtiles), blk, gemm_lds, stream>>>(
      x16, Wqkv16, buf_qkv, nullptr, ROWS, THREEC, CH, 1, ssf_scale1, ssf_shift1, nullptr);
  // 2) LoRA down (f16 out)
  gemm_wmma_f16<<<dim3(1, mtiles), blk, gemm_lds, stream>>>(
      x16, Wa16, nullptr, t_a16, ROWS, 64, CH, 0, nullptr, nullptr, nullptr);
  // 3) adapter down + exact GELU (f16 out)
  gemm_wmma_f16<<<dim3(1, mtiles), blk, gemm_lds, stream>>>(
      x16, Wadd16, nullptr, t_ad16, ROWS, 64, CH, 2, bad_down, nullptr, nullptr);
  // 4) router + sigmoid
  gemm_wmma_f16<<<dim3(1, mtiles), blk, gemm_lds, stream>>>(
      x16, Wrt16, router, nullptr, ROWS, NROUT, CH, 3, brouter, nullptr, nullptr);
  // 5) LoRA up, router-gated accumulate into buf_qkv
  gemm_wmma_f16<<<dim3(THREEC / 256, mtiles), blk, gemm_lds, stream>>>(
      t_a16, Wb16, buf_qkv, nullptr, ROWS, THREEC, 64, 4, nullptr, nullptr, router);
  // 6) adapter up (+bias), router-gated accumulate into buf_qkv
  gemm_wmma_f16<<<dim3(THREEC / 256, mtiles), blk, gemm_lds, stream>>>(
      t_ad16, Wadu16, buf_qkv, nullptr, ROWS, THREEC, 64, 5, bad_up, nullptr, router);
  // 7) repack to f16 attention operands
  pack_qkv<<<dim3((ROWS * CH + 255) / 256), blk, 0, stream>>>(
      buf_qkv, q16, k16, vT16);
  // 8) attention (f16 out)
  attn_wmma<<<dim3(BATCH * NHEAD * (SEQ / 64)), dim3(128), 0, stream>>>(
      q16, k16, vT16, o16);
  // 9) output projection + bias + SSF2
  gemm_wmma_f16<<<dim3(CH / 256, mtiles), blk, gemm_lds, stream>>>(
      o16, Wproj16, out, nullptr, ROWS, CH, CH, 6, bproj, ssf_scale2, ssf_shift2);
}
